// GraphSAGE_10161892622801
// MI455X (gfx1250) — matrix operand order repaired and checked by reading, not compiled, not run on hardware
//
#include <hip/hip_runtime.h>

typedef __attribute__((ext_vector_type(16))) __bf16 v16bf;
typedef __attribute__((ext_vector_type(8)))  float  v8f;
typedef __attribute__((ext_vector_type(4)))  unsigned int v4u;
typedef __attribute__((ext_vector_type(8)))  int v8i;
typedef __attribute__((ext_vector_type(4)))  int v4i;

union FragBf { uint4 q[2]; v16bf v; };

__device__ __forceinline__ unsigned short f2bf(float f) {
  union { float f; unsigned u; } c; c.f = f;
  unsigned r = c.u + 0x7FFFu + ((c.u >> 16) & 1u);   // round-to-nearest-even
  return (unsigned short)(r >> 16);
}
__device__ __forceinline__ float bf2f(unsigned short h) {
  union { unsigned u; float f; } c; c.u = (unsigned)h << 16;
  return c.f;
}

// --------- TDM: async 2D bf16 tile load global -> LDS with pad-to-stride40 ----------
// tile: 128 rows x 32 bf16 cols, row stride 128 elements in memory.
// pad: 16 DWORDs data (32 bf16) then 4 DWORDs pad (8 bf16) => LDS row stride 40 bf16.
// tensor_dim1 = rows remaining => hardware zero-fill for the tail block.
__device__ __forceinline__ void tdm_tile_load(unsigned lds_byte_off,
                                              const unsigned short* gptr,
                                              unsigned rows_remaining) {
  unsigned long long ga = (unsigned long long)(const void*)gptr;
  v4u g0;
  g0[0] = 1u;                                              // count=1, user mode, no gather
  g0[1] = lds_byte_off;                                    // lds_addr
  g0[2] = (unsigned)ga;                                    // global_addr[31:0]
  g0[3] = (unsigned)((ga >> 32) & 0x01FFFFFFu) | (2u << 30); // addr[56:32], type=2
  v8i g1;
  g1[0] = (int)((1u << 16) | (1u << 20) | (3u << 22) | (3u << 25)); // 2B elems, pad_en, every 16dw, +4dw
  g1[1] = (int)(128u << 16);                               // tensor_dim0 = 128 (bits 79:48 lo)
  g1[2] = (int)((rows_remaining & 0xFFFFu) << 16);         // dim0 hi=0, tensor_dim1 lo
  g1[3] = (int)((rows_remaining >> 16) | (32u << 16));     // tensor_dim1 hi, tile_dim0 = 32
  g1[4] = 128;                                             // tile_dim1 = 128, tile_dim2 = 0
  g1[5] = 128;                                             // tensor_dim0_stride = 128
  g1[6] = 0;                                               // stride hi, tensor_dim1_stride lo
  g1[7] = 0;
  v4i gz4 = { 0, 0, 0, 0 };
  v8i gz8 = { 0, 0, 0, 0, 0, 0, 0, 0 };
  __builtin_amdgcn_tensor_load_to_lds(g0, g1, gz4, gz4, gz8, 0);
}

__device__ __forceinline__ unsigned lds_off32(const void* p) {
  return (unsigned)(unsigned long long)p;   // LDS aperture: addr[31:0] is LDS byte offset
}

// ---------------- CSR build ----------------

__global__ void zero_int_k(int* __restrict__ p, int n) {
  int i = blockIdx.x * blockDim.x + threadIdx.x;
  if (i < n) p[i] = 0;
}

__global__ void count_k(const int* __restrict__ dst, int* __restrict__ cnt, int nE) {
  int e = blockIdx.x * blockDim.x + threadIdx.x;
  if (e < nE) atomicAdd(&cnt[dst[e]], 1);
}

// single-block chunked exclusive scan: off[0..n] from cnt[0..n-1]
__global__ void scan_k(const int* __restrict__ cnt, int* __restrict__ off, int n) {
  __shared__ int buf[256];
  __shared__ int running;
  const int tid = threadIdx.x;
  if (tid == 0) running = 0;
  __syncthreads();
  for (int base = 0; base < n; base += 256) {
    int i = base + tid;
    int v = (i < n) ? cnt[i] : 0;
    buf[tid] = v;
    __syncthreads();
#pragma unroll
    for (int s = 1; s < 256; s <<= 1) {
      int add = (tid >= s) ? buf[tid - s] : 0;
      __syncthreads();
      buf[tid] += add;
      __syncthreads();
    }
    if (i < n) off[i] = running + buf[tid] - v;   // exclusive
    __syncthreads();
    if (tid == 0) running += buf[255];
    __syncthreads();
  }
  if (tid == 0) off[n] = running;
}

__global__ void copy_int_k(const int* __restrict__ in, int* __restrict__ out, int n) {
  int i = blockIdx.x * blockDim.x + threadIdx.x;
  if (i < n) out[i] = in[i];
}

__global__ void place_k(const int* __restrict__ src, const int* __restrict__ dst,
                        int* __restrict__ cur, int* __restrict__ adj, int nE) {
  int e = blockIdx.x * blockDim.x + threadIdx.x;
  if (e < nE) {
    int p = atomicAdd(&cur[dst[e]], 1);
    adj[p] = src[e];
  }
}

// ---------------- conversions ----------------

__global__ void f2bf_k(const float* __restrict__ in, unsigned short* __restrict__ out, int n) {
  int i = blockIdx.x * blockDim.x + threadIdx.x;
  if (i < n) out[i] = f2bf(in[i]);
}

// ---------------- pull aggregation: mean over in-neighbors ----------------
// wave per node; lane covers features [4*lane, 4*lane+4). Each neighbor row read
// is a fully coalesced 256B wave load (L2-resident). f32 accumulation, bf16 out.
__global__ void gather_k(const unsigned short* __restrict__ Hbf,   // [N][128] bf16
                         const int* __restrict__ off, const int* __restrict__ adj,
                         unsigned short* __restrict__ Hn,          // [N][128] bf16 mean
                         int n_nodes) {
  int t = blockIdx.x * blockDim.x + threadIdx.x;
  int node = t >> 5;
  int lane = t & 31;
  if (node >= n_nodes) return;
  int s0 = off[node], s1 = off[node + 1];
  float a0 = 0.f, a1 = 0.f, a2 = 0.f, a3 = 0.f;
  for (int i = s0; i < s1; ++i) {
    int s = adj[i];
    if (i + 1 < s1)
      __builtin_prefetch((const void*)(Hbf + (size_t)adj[i + 1] * 128), 0, 0);
    ushort4 hv = *(const ushort4*)(Hbf + (size_t)s * 128 + (lane << 2));
    a0 += bf2f(hv.x); a1 += bf2f(hv.y); a2 += bf2f(hv.z); a3 += bf2f(hv.w);
  }
  int deg = s1 - s0;
  float inv = 1.0f / (float)(deg > 0 ? deg : 1);
  *(ushort4*)(Hn + (size_t)node * 128 + (lane << 2)) =
      make_ushort4(f2bf(a0 * inv), f2bf(a1 * inv), f2bf(a2 * inv), f2bf(a3 * inv));
}

// ---------------- fused SAGE layer GEMM (bf16 in, f32 acc, TDM-fed) ----------------
// out = act( Abf @ Wself + Nbf @ Wneigh + b ); 8 waves, 128x128 tile, K = 4 x 32.
// A tiles are double-buffered in LDS and DMA'd by the Tensor Data Mover, prefetching
// the next K-step while the current one computes (one barrier per iteration).
#define WT_STRIDE 136   // 128 + 8 bf16 pad  -> conflict-free b128 ds reads
#define A_STRIDE  40    // 32 + 8 bf16 pad (produced by TDM pad feature)

__global__ void sage_gemm(const unsigned short* __restrict__ Abf,     // [N][128] bf16 self input
                          const unsigned short* __restrict__ Nbf,     // [N][128] bf16 neighbor mean
                          const unsigned short* __restrict__ Wself,   // [128][128] bf16 (k,n)
                          const unsigned short* __restrict__ Wneigh,  // [128][128] bf16 (k,n)
                          const float* __restrict__ bias,             // [128]
                          float* __restrict__ HoutF,                  // f32 out (final layer)
                          unsigned short* __restrict__ HoutB,         // bf16 out (hidden layers)
                          int n_nodes, int do_relu) {
  extern __shared__ unsigned short smem[];
  unsigned short* Wts = smem;                       // [128][WT_STRIDE] transposed: Wt[n][k]
  unsigned short* Wtn = Wts + 128 * WT_STRIDE;
  unsigned short* Abuf[2];
  unsigned short* Nbuf[2];
  Abuf[0] = Wtn + 128 * WT_STRIDE;                  // [128][A_STRIDE] each
  Nbuf[0] = Abuf[0] + 128 * A_STRIDE;
  Abuf[1] = Nbuf[0] + 128 * A_STRIDE;
  Nbuf[1] = Abuf[1] + 128 * A_STRIDE;

  const int t    = threadIdx.x;
  const int wave = t >> 5;
  const int lane = t & 31;
  const int hi   = lane >> 4;
  const int l16  = lane & 15;
  const int m0   = blockIdx.x * 128;
  const unsigned rem = (unsigned)(n_nodes - m0);    // rows remaining; TDM zero-fills beyond

  // stage weights transposed into LDS (once per block)
#pragma unroll 4
  for (int i = 0; i < 64; ++i) {
    int idx = t + i * 256;            // idx = k*128 + n
    int k = idx >> 7, n = idx & 127;
    Wts[n * WT_STRIDE + k] = Wself[idx];
    Wtn[n * WT_STRIDE + k] = Wneigh[idx];
  }

  // prologue: DMA K-step 0 tiles into buffer 0
  if (wave == 0) {
    tdm_tile_load(lds_off32(Abuf[0]), Abf + (size_t)m0 * 128, rem);
    tdm_tile_load(lds_off32(Nbuf[0]), Nbf + (size_t)m0 * 128, rem);
  }

  v8f zero = {};
  v8f acc[8];
#pragma unroll
  for (int nt = 0; nt < 8; ++nt) acc[nt] = zero;

  for (int ks = 0; ks < 4; ++ks) {
    const int k0 = ks * 32;
    const unsigned short* As  = Abuf[ks & 1];
    const unsigned short* Ans = Nbuf[ks & 1];

    if (wave == 0) __builtin_amdgcn_s_wait_tensorcnt(0);   // tile ks landed in LDS
    __syncthreads();                                       // release all waves; prev reads done
    if (wave == 0 && ks < 3) {                             // async prefetch of next K-step
      tdm_tile_load(lds_off32(Abuf[(ks + 1) & 1]), Abf + (size_t)m0 * 128 + k0 + 32, rem);
      tdm_tile_load(lds_off32(Nbuf[(ks + 1) & 1]), Nbf + (size_t)m0 * 128 + k0 + 32, rem);
    }

    // A fragments (ISA 16-bit A 16x32 layout: hi half-wave takes K=8..15,24..31)
    const int abase = (wave * 16 + l16) * A_STRIDE + (hi ? 8 : 0);
    FragBf aS, aN;
    {
      const uint4* pa = (const uint4*)(As + abase);
      aS.q[0] = pa[0]; aS.q[1] = pa[2];
      const uint4* pn = (const uint4*)(Ans + abase);
      aN.q[0] = pn[0]; aN.q[1] = pn[2];
    }

#pragma unroll
    for (int nt = 0; nt < 8; ++nt) {
      const int wbase = (nt * 16 + l16) * WT_STRIDE + k0 + hi * 8;
      FragBf bS, bN;
      const uint4* pws = (const uint4*)(Wts + wbase);
      bS.q[0] = pws[0]; bS.q[1] = pws[2];
      const uint4* pwn = (const uint4*)(Wtn + wbase);
      bN.q[0] = pwn[0]; bN.q[1] = pwn[2];
      acc[nt] = __builtin_amdgcn_wmma_f32_16x16x32_bf16(
          false, aS.v, false, bS.v, (short)0, acc[nt], false, false);
      acc[nt] = __builtin_amdgcn_wmma_f32_16x16x32_bf16(
          false, aN.v, false, bN.v, (short)0, acc[nt], false, false);
    }
  }

  // epilogue: bias + optional relu (C/D layout: M = hi*8 + j, N = l16)
#pragma unroll
  for (int nt = 0; nt < 8; ++nt) {
    int col = nt * 16 + l16;
    float b = bias[col];
#pragma unroll
    for (int j = 0; j < 8; ++j) {
      int row = m0 + wave * 16 + hi * 8 + j;
      if (row < n_nodes) {
        float v = acc[nt][j] + b;
        if (do_relu) v = fmaxf(v, 0.0f);
        if (HoutB) HoutB[(size_t)row * 128 + col] = f2bf(v);
        else       HoutF[(size_t)row * 128 + col] = v;
      }
    }
  }
}

// ---------------- FC head: logits[N,40] = h @ fc_w + fc_b ----------------
__global__ void fc_gemm(const float* __restrict__ Hin,
                        const unsigned short* __restrict__ Wfc,   // [128][40] bf16 (k,n)
                        const float* __restrict__ bias,           // [40]
                        float* __restrict__ Out,                  // [N][40]
                        int n_nodes) {
  extern __shared__ unsigned short smem[];
  unsigned short* Wt = smem;                    // [48][WT_STRIDE] transposed, cols 40..47 zero
  unsigned short* As = Wt + 48 * WT_STRIDE;     // [128][A_STRIDE]

  const int t    = threadIdx.x;
  const int wave = t >> 5;
  const int lane = t & 31;
  const int hi   = lane >> 4;
  const int l16  = lane & 15;
  const int m0   = blockIdx.x * 128;

#pragma unroll 4
  for (int i = 0; i < 24; ++i) {
    int idx = t + i * 256;            // idx = n*128 + k
    int n = idx >> 7, k = idx & 127;
    Wt[n * WT_STRIDE + k] = (n < 40) ? Wfc[k * 40 + n] : (unsigned short)0;
  }

  v8f zero = {};
  v8f acc[3];
#pragma unroll
  for (int nt = 0; nt < 3; ++nt) acc[nt] = zero;

  for (int ks = 0; ks < 4; ++ks) {
    const int k0 = ks * 32;
    __syncthreads();
#pragma unroll
    for (int i = 0; i < 4; ++i) {
      int lin = t + i * 256;
      int row = lin >> 3;
      int c4  = lin & 7;
      int grow = m0 + row;
      float4 vs = make_float4(0.f, 0.f, 0.f, 0.f);
      if (grow < n_nodes)
        vs = *(const float4*)(Hin + (size_t)grow * 128 + k0 + c4 * 4);
      *(ushort4*)(As + row * A_STRIDE + c4 * 4) =
          make_ushort4(f2bf(vs.x), f2bf(vs.y), f2bf(vs.z), f2bf(vs.w));
    }
    __syncthreads();

    const int abase = (wave * 16 + l16) * A_STRIDE + (hi ? 8 : 0);
    FragBf a;
    {
      const uint4* pa = (const uint4*)(As + abase);
      a.q[0] = pa[0]; a.q[1] = pa[2];
    }
#pragma unroll
    for (int nt = 0; nt < 3; ++nt) {
      const int wbase = (nt * 16 + l16) * WT_STRIDE + k0 + hi * 8;
      FragBf b;
      const uint4* pw = (const uint4*)(Wt + wbase);
      b.q[0] = pw[0]; b.q[1] = pw[2];
      acc[nt] = __builtin_amdgcn_wmma_f32_16x16x32_bf16(
          false, a.v, false, b.v, (short)0, acc[nt], false, false);
    }
  }

#pragma unroll
  for (int nt = 0; nt < 3; ++nt) {
    int col = nt * 16 + l16;
    if (col < 40) {
      float b = bias[col];
#pragma unroll
      for (int j = 0; j < 8; ++j) {
        int row = m0 + wave * 16 + hi * 8 + j;
        if (row < n_nodes)
          Out[(size_t)row * 40 + col] = acc[nt][j] + b;
      }
    }
  }
}

// ---------------- host-side orchestration ----------------

extern "C" void kernel_launch(void* const* d_in, const int* in_sizes, int n_in,
                              void* d_out, int out_size, void* d_ws, size_t ws_size,
                              hipStream_t stream) {
  (void)n_in; (void)out_size; (void)ws_size;
  const int N = in_sizes[0] / 128;
  const int E = in_sizes[1];

  const float* x   = (const float*)d_in[0];
  const int* src   = (const int*)d_in[1];
  const int* dst   = (const int*)d_in[2];
  const float* b0  = (const float*)d_in[5];
  const float* b1  = (const float*)d_in[8];
  const float* b2  = (const float*)d_in[11];
  const float* fcb = (const float*)d_in[13];

  char* ws = (char*)d_ws;
  unsigned short* buf0 = (unsigned short*)ws;  ws += (size_t)N * 128 * sizeof(unsigned short);
  unsigned short* buf1 = (unsigned short*)ws;  ws += (size_t)N * 128 * sizeof(unsigned short);
  unsigned short* bufN = (unsigned short*)ws;  ws += (size_t)N * 128 * sizeof(unsigned short);
  int* cnt = (int*)ws;  ws += (size_t)N * sizeof(int);
  int* off = (int*)ws;  ws += (size_t)(N + 4) * sizeof(int);
  int* cur = (int*)ws;  ws += (size_t)N * sizeof(int);
  int* adj = (int*)ws;  ws += (size_t)E * sizeof(int);
  unsigned short* wbf = (unsigned short*)ws;   // 6*16384 + 5120 bf16 weights

  unsigned short* wS[3] = { wbf + 0 * 16384, wbf + 2 * 16384, wbf + 4 * 16384 };
  unsigned short* wN[3] = { wbf + 1 * 16384, wbf + 3 * 16384, wbf + 5 * 16384 };
  unsigned short* wF    = wbf + 6 * 16384;

  const int nB   = (N + 255) / 256;
  const int eB   = (E + 255) / 256;
  const int ndB  = (N * 128 + 255) / 256;
  const int gatB = (int)(((long long)N * 32 + 255) / 256);
  const int gemmBlocks = (N + 127) / 128;
  const size_t sageLds = (size_t)(2 * 128 * WT_STRIDE + 4 * 128 * A_STRIDE) * sizeof(unsigned short);
  const size_t fcLds   = (size_t)(48 * WT_STRIDE + 128 * A_STRIDE) * sizeof(unsigned short);

  // ---- CSR build (once; reused by all 3 layers) ----
  zero_int_k<<<nB, 256, 0, stream>>>(cnt, N);
  count_k<<<eB, 256, 0, stream>>>(dst, cnt, E);
  scan_k<<<1, 256, 0, stream>>>(cnt, off, N);
  copy_int_k<<<nB, 256, 0, stream>>>(off, cur, N);
  place_k<<<eB, 256, 0, stream>>>(src, dst, cur, adj, E);

  // ---- weights + x to bf16 ----
  const int widx[3][2] = { {3, 4}, {6, 7}, {9, 10} };
  for (int l = 0; l < 3; ++l) {
    f2bf_k<<<64, 256, 0, stream>>>((const float*)d_in[widx[l][0]], wS[l], 16384);
    f2bf_k<<<64, 256, 0, stream>>>((const float*)d_in[widx[l][1]], wN[l], 16384);
  }
  f2bf_k<<<20, 256, 0, stream>>>((const float*)d_in[12], wF, 5120);
  f2bf_k<<<ndB, 256, 0, stream>>>(x, buf0, N * 128);

  float* logits = (float*)d_out;
  float* hOut   = (float*)d_out + (size_t)N * 40;   // second tuple element: final conv h

  // layer 0: buf0 -> buf1 (relu)
  gather_k<<<gatB, 256, 0, stream>>>(buf0, off, adj, bufN, N);
  sage_gemm<<<gemmBlocks, 256, sageLds, stream>>>(buf0, bufN, wS[0], wN[0], b0,
                                                  (float*)nullptr, buf1, N, 1);
  // layer 1: buf1 -> buf0 (relu)
  gather_k<<<gatB, 256, 0, stream>>>(buf1, off, adj, bufN, N);
  sage_gemm<<<gemmBlocks, 256, sageLds, stream>>>(buf1, bufN, wS[1], wN[1], b1,
                                                  (float*)nullptr, buf0, N, 1);
  // layer 2: buf0 -> f32 h region of d_out (no relu)
  gather_k<<<gatB, 256, 0, stream>>>(buf0, off, adj, bufN, N);
  sage_gemm<<<gemmBlocks, 256, sageLds, stream>>>(buf0, bufN, wS[2], wN[2], b2,
                                                  hOut, (unsigned short*)nullptr, N, 0);
  // fc head
  fc_gemm<<<gemmBlocks, 256, fcLds, stream>>>(hOut, wF, fcb, logits, N);
}
